// PointManifold_NNML_32298154066459
// MI455X (gfx1250) — matrix operand order repaired and checked by reading, not compiled, not run on hardware
//
#include <hip/hip_runtime.h>
#include <hip/hip_bf16.h>

typedef __attribute__((ext_vector_type(16))) __bf16 v16bf;
typedef __attribute__((ext_vector_type(8)))  float  v8f;

#define Bn   8
#define Np   1024
#define KNB  20
#define NKC  (Np*KNB)      // 20480 (n,k) columns per batch
#define CATC 512           // concat channels for conv5
#define EPSB 1e-5f
#define NEGINF (-3.4e38f)

__device__ __forceinline__ __bf16 f2bf(float f) {
  unsigned u = __builtin_bit_cast(unsigned, f);
  unsigned short h = (unsigned short)(u >> 16);   // truncate-round to bf16
  return __builtin_bit_cast(__bf16, h);
}
__device__ __forceinline__ float lrelu(float y) { return y >= 0.f ? y : 0.2f * y; }

// --- WMMA fragment loaders (wave32, contiguous-k order used for both A and B), all b128 loads -------
// A (16x32 bf16): lane row = lane&15; 16 contiguous k at k0 = ks*32 + 16*(lane>>4)
__device__ __forceinline__ v16bf load_a(const float* __restrict__ row, int k0) {
  const float4* p = (const float4*)(row + k0);
  const float4* q = (const float4*)(row + k0 + 8);
  float4 c0 = p[0], c1 = p[1], c2 = q[0], c3 = q[1];
  v16bf a;
  a[0]=f2bf(c0.x);  a[1]=f2bf(c0.y);  a[2]=f2bf(c0.z);  a[3]=f2bf(c0.w);
  a[4]=f2bf(c1.x);  a[5]=f2bf(c1.y);  a[6]=f2bf(c1.z);  a[7]=f2bf(c1.w);
  a[8]=f2bf(c2.x);  a[9]=f2bf(c2.y);  a[10]=f2bf(c2.z); a[11]=f2bf(c2.w);
  a[12]=f2bf(c3.x); a[13]=f2bf(c3.y); a[14]=f2bf(c3.z); a[15]=f2bf(c3.w);
  return a;
}
// B (32x16 bf16): lane col = lane&15; 16 contiguous k at k0 = ks*32 + 16*(lane>>4)
__device__ __forceinline__ v16bf load_b(const float* __restrict__ row, int k0) {
  const float4* p = (const float4*)(row + k0);
  float4 c0 = p[0], c1 = p[1], c2 = p[2], c3 = p[3];
  v16bf r;
  r[0]=f2bf(c0.x);  r[1]=f2bf(c0.y);  r[2]=f2bf(c0.z);  r[3]=f2bf(c0.w);
  r[4]=f2bf(c1.x);  r[5]=f2bf(c1.y);  r[6]=f2bf(c1.z);  r[7]=f2bf(c1.w);
  r[8]=f2bf(c2.x);  r[9]=f2bf(c2.y);  r[10]=f2bf(c2.z); r[11]=f2bf(c2.w);
  r[12]=f2bf(c3.x); r[13]=f2bf(c3.y); r[14]=f2bf(c3.z); r[15]=f2bf(c3.w);
  return r;
}
// Edge feature chunk: whole 16-wide chunk is either (nbr - ctr) or ctr; branchless cndmask.
__device__ __forceinline__ v16bf load_b_edge(const float* __restrict__ rown,
                                             const float* __restrict__ rownb,
                                             int cs, bool isdiff) {
  const float4* pn = (const float4*)(rown + cs);
  const float4* pb = (const float4*)(rownb + cs);
  v16bf r;
#pragma unroll
  for (int q = 0; q < 4; ++q) {
    float4 vn = pn[q], vb = pb[q];
    r[4*q+0] = f2bf(isdiff ? (vb.x - vn.x) : vn.x);
    r[4*q+1] = f2bf(isdiff ? (vb.y - vn.y) : vn.y);
    r[4*q+2] = f2bf(isdiff ? (vb.z - vn.z) : vn.z);
    r[4*q+3] = f2bf(isdiff ? (vb.w - vn.w) : vn.w);
  }
  return r;
}

// ---------------------------------------------------------------------------
// conv0: 2x2 conv + train-BN over (B,N) + lrelu, gated by coord. Writes the
// 9 real channels of point-major padded xcT (B*N, 32); pad channels zeroed.
// blockIdx 0..2 = z,y,x branches; blockIdx 3 copies x + zero-fill.
// channels: [x(0..2), x2d_x(3..4), x2d_y(5..6), x2d_z(7..8), 0-pad(9..31)]
// ---------------------------------------------------------------------------
__global__ void k_conv0(const float* __restrict__ x,
                        const float* __restrict__ w00, const float* __restrict__ g00, const float* __restrict__ b00,
                        const float* __restrict__ w01, const float* __restrict__ g01, const float* __restrict__ b01,
                        float* __restrict__ xcT) {
  const int BN = Bn * Np;
  int br = blockIdx.x;
  if (br == 3) {
    for (int i = threadIdx.x; i < BN; i += 256) {
      int b = i >> 10, n = i & (Np-1);
      float* row = xcT + (size_t)i * 32;
      row[0] = x[(size_t)(b*3+0)*Np + n];
      row[1] = x[(size_t)(b*3+1)*Np + n];
      row[2] = x[(size_t)(b*3+2)*Np + n];
#pragma unroll
      for (int c = 9; c < 32; ++c) row[c] = 0.f;
    }
    return;
  }
  int ic0, ic1, coord, oc0; const float *w, *g, *bb;
  if (br == 0)      { ic0=0; ic1=1; coord=2; oc0=7; w=w00; g=g00; bb=b00; }
  else if (br == 1) { ic0=0; ic1=2; coord=1; oc0=5; w=w01; g=g01; bb=b01; }
  else              { ic0=1; ic1=2; coord=0; oc0=3; w=w01; g=g01; bb=b01; } // faithful w0_1 reuse

  __shared__ float ss0[256], ss1[256], qq0[256], qq1[256];
  float s0=0.f, s1=0.f, q0=0.f, q1=0.f;
  for (int i = threadIdx.x; i < BN; i += 256) {
    int b = i >> 10, n = i & (Np-1);
    float a0 = x[((size_t)b*3+ic0)*Np + n], a1 = x[((size_t)b*3+ic1)*Np + n];
    float y0 = w[0]*a0 + w[1]*a1;
    float y1 = w[2]*a0 + w[3]*a1;
    s0 += y0; s1 += y1; q0 += y0*y0; q1 += y1*y1;
  }
  ss0[threadIdx.x]=s0; ss1[threadIdx.x]=s1; qq0[threadIdx.x]=q0; qq1[threadIdx.x]=q1;
  __syncthreads();
  for (int st = 128; st > 0; st >>= 1) {
    if (threadIdx.x < st) {
      ss0[threadIdx.x]+=ss0[threadIdx.x+st]; ss1[threadIdx.x]+=ss1[threadIdx.x+st];
      qq0[threadIdx.x]+=qq0[threadIdx.x+st]; qq1[threadIdx.x]+=qq1[threadIdx.x+st];
    }
    __syncthreads();
  }
  float m0 = ss0[0]/BN, m1 = ss1[0]/BN;
  float v0 = qq0[0]/BN - m0*m0, v1 = qq1[0]/BN - m1*m1;
  float sc0 = g[0]*rsqrtf(v0+EPSB), sc1 = g[1]*rsqrtf(v1+EPSB);
  for (int i = threadIdx.x; i < BN; i += 256) {
    int b = i >> 10, n = i & (Np-1);
    float a0 = x[((size_t)b*3+ic0)*Np + n], a1 = x[((size_t)b*3+ic1)*Np + n];
    float cv = x[((size_t)b*3+coord)*Np + n];
    float* row = xcT + (size_t)i * 32;
    row[oc0  ] = lrelu((w[0]*a0 + w[1]*a1 - m0)*sc0 + bb[0]) * cv;
    row[oc0+1] = lrelu((w[2]*a0 + w[3]*a1 - m1)*sc1 + bb[1]) * cv;
  }
}

// ---------------------------------------------------------------------------
// pad weights (Cout x 2C) -> (Cout x 2CP): [w_diff | 0 | w_ctr | 0]
// ---------------------------------------------------------------------------
__global__ void k_packw(const float* __restrict__ w, float* __restrict__ wp,
                        int C, int CP, int Cout) {
  int K2p = 2*CP;
  int i = blockIdx.x*256 + threadIdx.x;
  if (i >= Cout*K2p) return;
  int co = i / K2p, k = i - co*K2p;
  float v = 0.f;
  if (k < C)                       v = w[(size_t)co*2*C + k];
  else if (k >= CP && k < CP + C)  v = w[(size_t)co*2*C + (k - CP) + C];
  wp[i] = v;
}

// xx[b*N+n] = |x_n|^2 (pad channels are zero, so safe to sum all CP)
__global__ void k_sqnorm(const float* __restrict__ xT, int ld, int CP, float* __restrict__ xx) {
  int i = blockIdx.x*256 + threadIdx.x;
  if (i >= Bn*Np) return;
  const float* row = xT + (size_t)i * ld;
  float s = 0.f;
  for (int c = 0; c < CP; ++c) s += row[c]*row[c];
  xx[i] = s;
}

// ---------------------------------------------------------------------------
// neg_dist[b,n,m] = 2*<x_n,x_m> - xx[n] - xx[m]; bf16 WMMA, f32 accum.
// One wave per 16x16 tile; fully unrolled k-loop (CP/32 WMMAs).
// ---------------------------------------------------------------------------
template<int CP>
__global__ void k_dist(const float* __restrict__ xT, int ld,
                       const float* __restrict__ xx, float* __restrict__ dist) {
  int wave = (blockIdx.x*256 + threadIdx.x) >> 5;
  int lane = threadIdx.x & 31;
  int hi = lane >> 4, lo = lane & 15;
  int tc = wave & 63, tr = (wave >> 6) & 63, b = wave >> 12;
  int n0 = tr*16, m0 = tc*16;
  const float* rowA = xT + (size_t)(b*Np + n0 + lo) * ld;
  const float* rowB = xT + (size_t)(b*Np + m0 + lo) * ld;
  v8f acc = {};
#pragma unroll
  for (int ks = 0; ks < CP/32; ++ks) {
    v16bf a  = load_a(rowA, ks*32 + (hi << 4));
    v16bf bm = load_b(rowB, ks*32 + (hi << 4));
    acc = __builtin_amdgcn_wmma_f32_16x16x32_bf16(false, a, false, bm, (short)0, acc, false, false);
  }
  const float* xxb = xx + (size_t)b*Np;
  float xm = xxb[m0 + lo];
#pragma unroll
  for (int v = 0; v < 8; ++v) {
    int n = n0 + v + (hi << 3);
    dist[(size_t)b*Np*Np + (size_t)n*Np + m0 + lo] = 2.f*acc[v] - xxb[n] - xm;
  }
}

// ---------------------------------------------------------------------------
// top-20 per row; one wave per row, row cached in LDS.
// ---------------------------------------------------------------------------
__global__ void k_topk(const float* __restrict__ dist, int* __restrict__ idxout) {
  __shared__ float rows[8][Np];
  int wid = threadIdx.x >> 5, lane = threadIdx.x & 31;
  int row = blockIdx.x*8 + wid;             // [0, B*N)
  const float* src = dist + (size_t)row * Np;
  float* r = rows[wid];
  for (int i = lane; i < Np; i += 32) r[i] = src[i];
  __syncthreads();
  for (int it = 0; it < KNB; ++it) {
    float bv = NEGINF; int bi = 0x7fffffff;
    for (int i = lane; i < Np; i += 32) {
      float v = r[i];
      if (v > bv || (v == bv && i < bi)) { bv = v; bi = i; }
    }
#pragma unroll
    for (int s = 16; s > 0; s >>= 1) {
      float ov = __shfl_xor(bv, s, 32);
      int   oi = __shfl_xor(bi, s, 32);
      if (ov > bv || (ov == bv && oi < bi)) { bv = ov; bi = oi; }
    }
    if (lane == 0) { idxout[row*KNB + it] = bi; r[bi] = NEGINF; }
    __syncthreads();
  }
}

// ---------------------------------------------------------------------------
// EdgeConv GEMM: h[b,co,col] = sum_k wpad[co,k]*feat[k,col],
// feat = [x_nbr - x_ctr (CP), x_ctr (CP)] gathered on the fly, branchless.
// ---------------------------------------------------------------------------
template<int CP, int Cout>
__global__ void k_edge_gemm(const float* __restrict__ xT, int ld,
                            const int* __restrict__ idx,
                            const float* __restrict__ wp,
                            float* __restrict__ hout) {
  int wave = (blockIdx.x*256 + threadIdx.x) >> 5;
  int lane = threadIdx.x & 31;
  int hi = lane >> 4, lo = lane & 15;
  const int ctiles = Cout >> 4;
  int ct = wave % ctiles;
  int rest = wave / ctiles;
  int colTile = rest % (NKC/16);
  int b = rest / (NKC/16);
  int col = colTile*16 + lo;
  int n = col / KNB, kk = col - n*KNB;
  int nbr = idx[((size_t)b*Np + n)*KNB + kk];
  const float* rown  = xT + (size_t)(b*Np + n)   * ld;
  const float* rownb = xT + (size_t)(b*Np + nbr) * ld;
  const float* wrow  = wp + (size_t)(ct*16 + lo) * (2*CP);
  v8f acc = {};
#pragma unroll
  for (int ks = 0; ks < (2*CP)/32; ++ks) {
    v16bf a = load_a(wrow, ks*32 + (hi << 4));
    int kstart = ks*32 + (hi << 4);
    bool isdiff = kstart < CP;               // whole 16-chunk on one side
    int cs = isdiff ? kstart : kstart - CP;
    v16bf bm = load_b_edge(rown, rownb, cs, isdiff);
    acc = __builtin_amdgcn_wmma_f32_16x16x32_bf16(false, a, false, bm, (short)0, acc, false, false);
  }
  size_t base = (size_t)b*Cout*NKC;
#pragma unroll
  for (int v = 0; v < 8; ++v) {
    int co = ct*16 + v + (hi << 3);
    hout[base + (size_t)co*NKC + col] = acc[v];
  }
}

// ---------------------------------------------------------------------------
// per-channel mean/var over (B, M), layout h[(b*Cper + c)*M + j]
// ---------------------------------------------------------------------------
__global__ void k_stats(const float* __restrict__ h, int Cper, int M,
                        float* __restrict__ mean, float* __restrict__ var) {
  int c = blockIdx.x;
  __shared__ float ss[256], sq[256];
  float s = 0.f, q = 0.f;
  int total = Bn * M;
  for (int i = threadIdx.x; i < total; i += 256) {
    int b = i / M, j = i - b*M;
    float v = h[((size_t)b*Cper + c)*(size_t)M + j];
    s += v; q += v*v;
  }
  ss[threadIdx.x] = s; sq[threadIdx.x] = q;
  __syncthreads();
  for (int st = 128; st > 0; st >>= 1) {
    if (threadIdx.x < st) { ss[threadIdx.x] += ss[threadIdx.x+st]; sq[threadIdx.x] += sq[threadIdx.x+st]; }
    __syncthreads();
  }
  if (threadIdx.x == 0) {
    float m = ss[0] / total;
    mean[c] = m;
    var[c] = sq[0] / total - m*m;
  }
}

// BN + lrelu + max over k -> transposed slice outT[(b*N+n)*ldo + c]
__global__ void k_bn_maxT(const float* __restrict__ h, const float* __restrict__ mean,
                          const float* __restrict__ var, const float* __restrict__ g,
                          const float* __restrict__ beta,
                          float* __restrict__ outT, int Cout, int ldo) {
  int i = blockIdx.x*256 + threadIdx.x;           // B*Cout*N
  int n = i & (Np-1);
  int c = (i >> 10) % Cout;
  int b = i / (Cout*Np);
  float sc = g[c] * rsqrtf(var[c] + EPSB);
  float sh = beta[c] - mean[c]*sc;
  const float* hp = h + ((size_t)b*Cout + c)*NKC + (size_t)n*KNB;
  float m = NEGINF;
  for (int kk = 0; kk < KNB; ++kk) m = fmaxf(m, lrelu(hp[kk]*sc + sh));
  outT[(size_t)(b*Np + n)*ldo + c] = m;
}

// ---------------------------------------------------------------------------
// conv5: 512 -> 1024 over (B,N); B-operand contiguous rows of catT. WMMA x16.
// ---------------------------------------------------------------------------
__global__ void k_conv5(const float* __restrict__ catT, const float* __restrict__ w5,
                        float* __restrict__ hout) {
  int wave = (blockIdx.x*256 + threadIdx.x) >> 5;
  int lane = threadIdx.x & 31;
  int hi = lane >> 4, lo = lane & 15;
  int ct = wave & 63;            // 1024/16 output-channel tiles
  int nt = (wave >> 6) & 63;     // 1024/16 column tiles
  int b  = wave >> 12;
  const float* rowB = catT + (size_t)(b*Np + nt*16 + lo) * CATC;
  const float* rowA = w5   + (size_t)(ct*16 + lo) * CATC;
  v8f acc = {};
#pragma unroll
  for (int ks = 0; ks < CATC/32; ++ks) {
    v16bf a  = load_a(rowA, ks*32 + (hi << 4));
    v16bf bm = load_b(rowB, ks*32 + (hi << 4));
    acc = __builtin_amdgcn_wmma_f32_16x16x32_bf16(false, a, false, bm, (short)0, acc, false, false);
  }
  int n = nt*16 + lo;
#pragma unroll
  for (int v = 0; v < 8; ++v) {
    int o = ct*16 + v + (hi << 3);
    hout[((size_t)b*1024 + o)*Np + n] = acc[v];
  }
}

__global__ void k_bn_lrelu(const float* __restrict__ h, const float* __restrict__ mean,
                           const float* __restrict__ var, const float* __restrict__ g,
                           const float* __restrict__ beta, float* __restrict__ out,
                           int Cper, int M) {
  size_t i = (size_t)blockIdx.x*256 + threadIdx.x;
  size_t total = (size_t)Bn*Cper*M;
  if (i >= total) return;
  int c = (int)((i / (size_t)M) % Cper);
  float sc = g[c]*rsqrtf(var[c]+EPSB);
  out[i] = lrelu((h[i]-mean[c])*sc + beta[c]);
}

__global__ void k_pool(const float* __restrict__ h, float* __restrict__ pooled) {
  int i = blockIdx.x*256 + threadIdx.x;          // B*1024
  if (i >= Bn*1024) return;
  int b = i >> 10, c = i & 1023;
  const float* p = h + ((size_t)b*1024 + c)*Np;
  float m = NEGINF, s = 0.f;
  for (int n = 0; n < Np; ++n) { float v = p[n]; m = fmaxf(m, v); s += v; }
  pooled[b*2048 + c]        = m;
  pooled[b*2048 + 1024 + c] = s * (1.f/Np);
}

__global__ void k_fc(const float* __restrict__ in, const float* __restrict__ w,
                     const float* __restrict__ bias, float* __restrict__ out,
                     int IN, int OUT) {
  int i = blockIdx.x*256 + threadIdx.x;
  if (i >= Bn*OUT) return;
  int b = i / OUT, o = i - b*OUT;
  float s = bias ? bias[o] : 0.f;
  const float* ip = in + (size_t)b*IN;
  const float* wp = w + (size_t)o*IN;
  for (int k = 0; k < IN; ++k) s += ip[k]*wp[k];
  out[i] = s;
}

__global__ void k_bn_fc(const float* __restrict__ y, const float* __restrict__ g,
                        const float* __restrict__ beta, float* __restrict__ out, int OUT) {
  int o = blockIdx.x*256 + threadIdx.x;
  if (o >= OUT) return;
  float s = 0.f, q = 0.f;
  for (int b = 0; b < Bn; ++b) { float v = y[b*OUT+o]; s += v; q += v*v; }
  float m = s/Bn, vv = q/Bn - m*m;
  float sc = g[o]*rsqrtf(vv+EPSB);
  for (int b = 0; b < Bn; ++b)
    out[b*OUT+o] = lrelu((y[b*OUT+o]-m)*sc + beta[o]);
}

// ---------------------------------------------------------------------------
extern "C" void kernel_launch(void* const* d_in, const int* in_sizes, int n_in,
                              void* d_out, int out_size, void* d_ws, size_t ws_size,
                              hipStream_t stream) {
  const float* x    = (const float*)d_in[0];
  const float* w00  = (const float*)d_in[1];
  const float* g00  = (const float*)d_in[2];
  const float* b00  = (const float*)d_in[3];
  const float* w01  = (const float*)d_in[4];
  const float* g01  = (const float*)d_in[5];
  const float* b01  = (const float*)d_in[6];
  const float* w1 = (const float*)d_in[7];  const float* g1 = (const float*)d_in[8];  const float* bb1 = (const float*)d_in[9];
  const float* w2 = (const float*)d_in[10]; const float* g2 = (const float*)d_in[11]; const float* bb2 = (const float*)d_in[12];
  const float* w3 = (const float*)d_in[13]; const float* g3 = (const float*)d_in[14]; const float* bb3 = (const float*)d_in[15];
  const float* w4 = (const float*)d_in[16]; const float* g4 = (const float*)d_in[17]; const float* bb4 = (const float*)d_in[18];
  const float* w5 = (const float*)d_in[19]; const float* g5 = (const float*)d_in[20]; const float* bb5 = (const float*)d_in[21];
  const float* w6 = (const float*)d_in[22]; const float* g6 = (const float*)d_in[23]; const float* bb6 = (const float*)d_in[24];
  const float* w7 = (const float*)d_in[25]; const float* bias7 = (const float*)d_in[26];
  const float* g7 = (const float*)d_in[27]; const float* bb7 = (const float*)d_in[28];
  const float* w8 = (const float*)d_in[29]; const float* bias8 = (const float*)d_in[30];

  char* ws = (char*)d_ws;
  size_t off = 0;
  auto take = [&](size_t bytes) -> void* {
    off = (off + 255) & ~(size_t)255;
    void* p = ws + off;
    off += bytes;
    return p;
  };
  float* xcT   = (float*)take((size_t)Bn*Np*32*4);      // (B*N, 32) padded
  float* catT  = (float*)take((size_t)Bn*Np*CATC*4);    // (B*N, 512): x1|x2|x3|x4
  float* wpad  = (float*)take((size_t)256*512*4);
  float* xx    = (float*)take((size_t)Bn*Np*4);
  float* dist  = (float*)take((size_t)Bn*Np*Np*4);
  int*   idx   = (int*)  take((size_t)Bn*Np*KNB*4);
  float* hraw  = (float*)take((size_t)Bn*256*NKC*4);    // reused per block + conv5 raw
  float* h5act = (float*)take((size_t)Bn*1024*Np*4);
  float* meanb = (float*)take(1024*4);
  float* varb  = (float*)take(1024*4);
  float* pooled= (float*)take((size_t)Bn*2048*4);
  float* fc6y  = (float*)take((size_t)Bn*512*4);
  float* fc6a  = (float*)take((size_t)Bn*512*4);
  float* fc7y  = (float*)take((size_t)Bn*256*4);
  float* fc7a  = (float*)take((size_t)Bn*256*4);

  // conv0 branches + assemble padded transposed xc (B*N, 32)
  k_conv0<<<4, 256, 0, stream>>>(x, w00, g00, b00, w01, g01, b01, xcT);

  const int distBlocks = (Bn*64*64)/8;
  const int topkBlocks = (Bn*Np)/8;

  // ---- edge block 1: in xcT (CP=32), out catT[:,0:64] ----
  k_packw<<<(64*64+255)/256, 256, 0, stream>>>(w1, wpad, 9, 32, 64);
  k_sqnorm<<<(Bn*Np)/256, 256, 0, stream>>>(xcT, 32, 32, xx);
  k_dist<32><<<distBlocks, 256, 0, stream>>>(xcT, 32, xx, dist);
  k_topk<<<topkBlocks, 256, 0, stream>>>(dist, idx);
  k_edge_gemm<32,64><<<(Bn*4*(NKC/16))/8, 256, 0, stream>>>(xcT, 32, idx, wpad, hraw);
  k_stats<<<64, 256, 0, stream>>>(hraw, 64, NKC, meanb, varb);
  k_bn_maxT<<<(Bn*64*Np)/256, 256, 0, stream>>>(hraw, meanb, varb, g1, bb1, catT + 0, 64, CATC);

  // ---- edge block 2: in catT[:,0:64], out catT[:,64:128] ----
  k_packw<<<(64*128+255)/256, 256, 0, stream>>>(w2, wpad, 64, 64, 64);
  k_sqnorm<<<(Bn*Np)/256, 256, 0, stream>>>(catT + 0, CATC, 64, xx);
  k_dist<64><<<distBlocks, 256, 0, stream>>>(catT + 0, CATC, xx, dist);
  k_topk<<<topkBlocks, 256, 0, stream>>>(dist, idx);
  k_edge_gemm<64,64><<<(Bn*4*(NKC/16))/8, 256, 0, stream>>>(catT + 0, CATC, idx, wpad, hraw);
  k_stats<<<64, 256, 0, stream>>>(hraw, 64, NKC, meanb, varb);
  k_bn_maxT<<<(Bn*64*Np)/256, 256, 0, stream>>>(hraw, meanb, varb, g2, bb2, catT + 64, 64, CATC);

  // ---- edge block 3: in catT[:,64:128], out catT[:,128:256] ----
  k_packw<<<(128*128+255)/256, 256, 0, stream>>>(w3, wpad, 64, 64, 128);
  k_sqnorm<<<(Bn*Np)/256, 256, 0, stream>>>(catT + 64, CATC, 64, xx);
  k_dist<64><<<distBlocks, 256, 0, stream>>>(catT + 64, CATC, xx, dist);
  k_topk<<<topkBlocks, 256, 0, stream>>>(dist, idx);
  k_edge_gemm<64,128><<<(Bn*8*(NKC/16))/8, 256, 0, stream>>>(catT + 64, CATC, idx, wpad, hraw);
  k_stats<<<128, 256, 0, stream>>>(hraw, 128, NKC, meanb, varb);
  k_bn_maxT<<<(Bn*128*Np)/256, 256, 0, stream>>>(hraw, meanb, varb, g3, bb3, catT + 128, 128, CATC);

  // ---- edge block 4: in catT[:,128:256], out catT[:,256:512] ----
  k_packw<<<(256*256+255)/256, 256, 0, stream>>>(w4, wpad, 128, 128, 256);
  k_sqnorm<<<(Bn*Np)/256, 256, 0, stream>>>(catT + 128, CATC, 128, xx);
  k_dist<128><<<distBlocks, 256, 0, stream>>>(catT + 128, CATC, xx, dist);
  k_topk<<<topkBlocks, 256, 0, stream>>>(dist, idx);
  k_edge_gemm<128,256><<<(Bn*16*(NKC/16))/8, 256, 0, stream>>>(catT + 128, CATC, idx, wpad, hraw);
  k_stats<<<256, 256, 0, stream>>>(hraw, 256, NKC, meanb, varb);
  k_bn_maxT<<<(Bn*256*Np)/256, 256, 0, stream>>>(hraw, meanb, varb, g4, bb4, catT + 256, 256, CATC);

  // ---- conv5 (512 -> 1024) + BN + lrelu ----
  k_conv5<<<(Bn*64*64)/8, 256, 0, stream>>>(catT, w5, hraw);
  k_stats<<<1024, 256, 0, stream>>>(hraw, 1024, Np, meanb, varb);
  k_bn_lrelu<<<(Bn*1024*Np)/256, 256, 0, stream>>>(hraw, meanb, varb, g5, bb5, h5act, 1024, Np);

  // ---- pooling + FC head ----
  k_pool<<<(Bn*1024 + 255)/256, 256, 0, stream>>>(h5act, pooled);
  k_fc<<<(Bn*512 + 255)/256, 256, 0, stream>>>(pooled, w6, nullptr, fc6y, 2048, 512);
  k_bn_fc<<<2, 256, 0, stream>>>(fc6y, g6, bb6, fc6a, 512);
  k_fc<<<(Bn*256 + 255)/256, 256, 0, stream>>>(fc6a, w7, bias7, fc7y, 512, 256);
  k_bn_fc<<<1, 256, 0, stream>>>(fc7y, g7, bb7, fc7a, 256);
  k_fc<<<(Bn*40 + 255)/256, 256, 0, stream>>>(fc7a, w8, bias8, (float*)d_out, 256, 40);
}
